// AttnTalkingHead_16947940950093
// MI455X (gfx1250) — matrix operand order repaired and checked by reading, compiled, not run on hardware
//
#include <hip/hip_runtime.h>
#include <stdint.h>

typedef __bf16 bf16_t;
typedef __attribute__((ext_vector_type(16))) __bf16 v16bf;
typedef __attribute__((ext_vector_type(8)))  __bf16 v8bf;
typedef __attribute__((ext_vector_type(4)))  __bf16 v4bf;
typedef __attribute__((ext_vector_type(8)))  float  v8f;

#define B_ 4
#define S_ 1024
#define D_ 768
#define H_ 12
#define E_ 64
#define BH 48        // B_*H_
#define BS 4096      // B_*S_
#define D3 2304      // 3*D_

// ---------- TDM support detection ----------
#if defined(__has_builtin)
#  if __has_builtin(__builtin_amdgcn_tensor_load_to_lds)
#    define HAVE_TDM 1
#  endif
#endif

#ifdef HAVE_TDM
typedef __attribute__((ext_vector_type(4))) unsigned int u32x4;
typedef __attribute__((ext_vector_type(8))) int          i32x8;
typedef __attribute__((ext_vector_type(4))) int          i32x4;

// 2D tile load via Tensor Data Mover: dim0 elements/row (bf16), `rows` rows,
// row stride `stride` elements. D# bit layout per CDNA5 ISA ch.8.
__device__ __forceinline__ void tdm_load_2d(uint32_t lds_off, uint64_t gaddr,
                                            uint32_t dim0, uint32_t rows,
                                            uint32_t stride) {
  u32x4 g0;
  g0[0] = 1u;                                   // count=1, user descriptor
  g0[1] = lds_off;                              // lds_addr (byte offset)
  g0[2] = (uint32_t)gaddr;                      // global_addr[31:0]
  g0[3] = ((uint32_t)(gaddr >> 32) & 0x01FFFFFFu) | (2u << 30);  // addr[56:32] | type=2
  i32x8 g1;
  g1[0] = 1 << 16;                              // data_size=1 (2 bytes/elem)
  g1[1] = (int)(dim0 << 16);                    // tensor_dim0[15:0] @ bits[63:48]
  g1[2] = (int)(rows << 16);                    // tensor_dim1[15:0] @ bits[95:80]
  g1[3] = (int)(dim0 << 16);                    // tile_dim0 @ bits[127:112]
  g1[4] = (int)rows;                            // tile_dim1 @ bits[143:128]
  g1[5] = (int)stride;                          // tensor_dim0_stride[31:0]
  g1[6] = 0;
  g1[7] = 0;
  i32x4 z4 = {0, 0, 0, 0};
#if __clang_major__ >= 23
  i32x8 z8 = {0, 0, 0, 0, 0, 0, 0, 0};
  __builtin_amdgcn_tensor_load_to_lds(g0, g1, z4, z4, z8, 0);
#else
  __builtin_amdgcn_tensor_load_to_lds(g0, g1, z4, z4, 0);
#endif
}

__device__ __forceinline__ void tdm_wait0() {
#if __has_builtin(__builtin_amdgcn_s_wait_tensorcnt)
  __builtin_amdgcn_s_wait_tensorcnt(0);
#else
  asm volatile("s_wait_tensorcnt 0x0" ::: "memory");
#endif
}
#endif  // HAVE_TDM

// ---------- helpers ----------

__device__ __forceinline__ bf16_t f2bf(float f) {
  union { float f; uint32_t u; } x; x.f = f;
  uint32_t r = x.u + 0x7FFFu + ((x.u >> 16) & 1u);   // round-to-nearest-even
  union { uint16_t u; bf16_t b; } y; y.u = (uint16_t)(r >> 16);
  return y.b;
}

// A fragment (16x32 bf16, row-major source, leading dim lda):
// lanes 0-15 hold row (lane&15), K=0..15;
// lanes 16-31 hold the same rows, K=16..31.
__device__ __forceinline__ v16bf load_a_frag(const bf16_t* __restrict__ base,
                                             int lda, int lane, int kk) {
  const int r  = lane & 15;
  const int kh = (lane >> 4) << 4;
  const bf16_t* p = base + (size_t)r * lda + kk + kh;
  union { v16bf v; v8bf h[2]; } u;
  u.h[0] = *(const v8bf*)(p);
  u.h[1] = *(const v8bf*)(p + 8);
  return u.v;
}

// B fragment (32x16 bf16): lane holds column (lane&15); lanes 0-15 K=0..15,
// lanes 16-31 K=16..31 (contiguous 32-byte run per lane). Source is the
// row-major matrix whose ROWS are the needed columns (A @ Bt^T).
__device__ __forceinline__ v16bf load_b_frag(const bf16_t* __restrict__ base,
                                             int ldb, int lane, int kk) {
  const int c  = lane & 15;
  const int kh = (lane >> 4) << 4;
  const bf16_t* p = base + (size_t)c * ldb + kk + kh;
  union { v16bf v; v8bf h[2]; } u;
  u.h[0] = *(const v8bf*)(p);
  u.h[1] = *(const v8bf*)(p + 8);
  return u.v;
}

__device__ __forceinline__ v8f wmma_bf16(v16bf a, v16bf b, v8f c) {
  return __builtin_amdgcn_wmma_f32_16x16x32_bf16(false, a, false, b,
                                                 (short)0, c, false, false);
}

// ---------- K0: fp32 -> bf16 ----------

__global__ __launch_bounds__(256) void convert_kernel(const float* __restrict__ in,
                                                      bf16_t* __restrict__ out, int n) {
  int i = blockIdx.x * 256 + threadIdx.x;
  if (i < n) out[i] = f2bf(in[i]);
}

// ---------- K1: QKV projection ----------
// out[s,f] = sum_d x[s,d]*Wqkv[f,d]; f = c*768 + h*64 + e. One wave = 16 rows x 64 cols
// (= full E of one (c,h)). jobs = 256 m-tiles x 36 (c,h)-groups = 9216.

__global__ __launch_bounds__(128) void qkv_gemm_kernel(
    const bf16_t* __restrict__ xb, const bf16_t* __restrict__ wb,
    bf16_t* __restrict__ qb, bf16_t* __restrict__ kb, bf16_t* __restrict__ vb) {
  const int lane = threadIdx.x & 31;
  const int job  = blockIdx.x * 4 + (threadIdx.x >> 5);
  const int tm = job & 255;
  const int g  = job >> 8;            // 0..35
  const int c  = g / 12;
  const int h  = g % 12;
  const bf16_t* Abase = xb + (size_t)(tm * 16) * D_;
  const bf16_t* Bbase = wb + (size_t)(c * D_ + h * E_) * D_;
  v8f acc[4] = {};
  for (int kk = 0; kk < D_; kk += 32) {
    v16bf a  = load_a_frag(Abase, D_, lane, kk);
    v16bf b0 = load_b_frag(Bbase +  0 * D_, D_, lane, kk);
    v16bf b1 = load_b_frag(Bbase + 16 * D_, D_, lane, kk);
    v16bf b2 = load_b_frag(Bbase + 32 * D_, D_, lane, kk);
    v16bf b3 = load_b_frag(Bbase + 48 * D_, D_, lane, kk);
    acc[0] = wmma_bf16(a, b0, acc[0]);
    acc[1] = wmma_bf16(a, b1, acc[1]);
    acc[2] = wmma_bf16(a, b2, acc[2]);
    acc[3] = wmma_bf16(a, b3, acc[3]);
  }
  bf16_t* outc = (c == 0) ? qb : (c == 1) ? kb : vb;
  const int n  = lane & 15;
  const int mh = (lane >> 4) << 3;
#pragma unroll
  for (int i = 0; i < 4; ++i)
#pragma unroll
    for (int r = 0; r < 8; ++r) {
      int m = tm * 16 + mh + r;
      int b = m >> 10, s = m & 1023;
      outc[((((size_t)b * H_ + h) * S_ + s) << 6) + i * 16 + n] = f2bf(acc[i][r]);
    }
}

// ---------- K2: attn = q @ k^T per (b,h) ----------
// Block = 4 waves sharing one 64-row k-panel (TDM -> 8KB LDS), covering 64(q)x64(k).
// blocks = 48 bh x 16 k-panels x 16 q-groups = 12288.

__global__ __launch_bounds__(128) void attn_gemm_kernel(
    const bf16_t* __restrict__ qb, const bf16_t* __restrict__ kb,
    float* __restrict__ attn) {
  __shared__ bf16_t kpanel[64 * 64];            // 8 KB
  const int lane = threadIdx.x & 31;
  const int wid  = threadIdx.x >> 5;
  const int bh   = blockIdx.x >> 8;
  const int rest = blockIdx.x & 255;
  const int tk4  = rest & 15;
  const int tq   = (rest >> 4) * 4 + wid;
  const bf16_t* Bsrc = kb + ((size_t)bh * S_ + tk4 * 64) * E_;

#ifdef HAVE_TDM
  if (wid == 0) {
    tdm_load_2d((uint32_t)(uintptr_t)(void*)kpanel,
                (uint64_t)(uintptr_t)(const void*)Bsrc,
                /*dim0=*/E_, /*rows=*/64, /*stride=*/E_);
    tdm_wait0();
  }
  __syncthreads();
#else
  {   // cooperative copy fallback: 8KB contiguous = 512 x b128
    const v8bf* src = (const v8bf*)Bsrc;
    v8bf* dst = (v8bf*)kpanel;
    for (int i = threadIdx.x; i < 512; i += 128) dst[i] = src[i];
    __syncthreads();
  }
#endif

  const bf16_t* Abase = qb + ((size_t)bh * S_ + tq * 16) * E_;
  v8f acc[4] = {};
  for (int kk = 0; kk < E_; kk += 32) {
    v16bf a  = load_a_frag(Abase, E_, lane, kk);
    v16bf b0 = load_b_frag(kpanel +  0 * E_, E_, lane, kk);
    v16bf b1 = load_b_frag(kpanel + 16 * E_, E_, lane, kk);
    v16bf b2 = load_b_frag(kpanel + 32 * E_, E_, lane, kk);
    v16bf b3 = load_b_frag(kpanel + 48 * E_, E_, lane, kk);
    acc[0] = wmma_bf16(a, b0, acc[0]);
    acc[1] = wmma_bf16(a, b1, acc[1]);
    acc[2] = wmma_bf16(a, b2, acc[2]);
    acc[3] = wmma_bf16(a, b3, acc[3]);
  }
  const int n = lane & 15, mh = (lane >> 4) << 3;
  float* out = attn + (size_t)bh * S_ * S_;
#pragma unroll
  for (int i = 0; i < 4; ++i)
#pragma unroll
    for (int r = 0; r < 8; ++r) {
      int qq = tq * 16 + mh + r;
      out[(size_t)qq * S_ + tk4 * 64 + i * 16 + n] = acc[i][r];
    }
}

// ---------- K3: fused talking-heads mix -> softmax -> mix ----------
// One block per (b,q) row: stage attn[b,:,q,:] (12x1024 f32 = 48KB) in LDS,
// mix over heads with Wl+bl, softmax per head over k, mix with Ww+bw, emit bf16.

__global__ __launch_bounds__(256) void th_softmax_kernel(
    const float* __restrict__ attn,
    const float* __restrict__ Wl, const float* __restrict__ bl,
    const float* __restrict__ Ww, const float* __restrict__ bw,
    bf16_t* __restrict__ attn2) {
  __shared__ float rowbuf[H_ * S_];      // 48 KB
  __shared__ float red[H_ * 256];        // 12 KB reduction scratch
  __shared__ float wl_s[144], ww_s[144], bl_s[12], bw_s[12];
  const int tid = threadIdx.x;
  const int bq  = blockIdx.x;
  const int b = bq >> 10, qrow = bq & 1023;

  if (tid < 144) { wl_s[tid] = Wl[tid]; ww_s[tid] = Ww[tid]; }
  if (tid < 12)  { bl_s[tid] = bl[tid]; bw_s[tid] = bw[tid]; }

  const size_t base = ((size_t)b * H_) * S_ * S_ + (size_t)qrow * S_;
#pragma unroll
  for (int h = 0; h < H_; ++h) {
    float4 vv = *(const float4*)(attn + base + (size_t)h * S_ * S_ + tid * 4);
    *(float4*)(rowbuf + h * S_ + tid * 4) = vv;
  }
  __syncthreads();

  // pre-softmax head mix (in place per-k) + per-head local max
  float lmax[H_];
#pragma unroll
  for (int g = 0; g < H_; ++g) lmax[g] = -3.0e38f;
  for (int j = 0; j < 4; ++j) {
    const int k = tid * 4 + j;
    float a[H_];
#pragma unroll
    for (int h = 0; h < H_; ++h) a[h] = rowbuf[h * S_ + k];
#pragma unroll
    for (int g = 0; g < H_; ++g) {
      float m = bl_s[g];
#pragma unroll
      for (int h = 0; h < H_; ++h) m = fmaf(a[h], wl_s[g * 12 + h], m);
      rowbuf[g * S_ + k] = m;
      lmax[g] = fmaxf(lmax[g], m);
    }
  }
#pragma unroll
  for (int g = 0; g < H_; ++g) red[g * 256 + tid] = lmax[g];
  for (int s2 = 128; s2 > 0; s2 >>= 1) {
    __syncthreads();
    if (tid < s2)
#pragma unroll
      for (int g = 0; g < H_; ++g)
        red[g * 256 + tid] = fmaxf(red[g * 256 + tid], red[g * 256 + tid + s2]);
  }
  __syncthreads();
  float rmax[H_];
#pragma unroll
  for (int g = 0; g < H_; ++g) rmax[g] = red[g * 256];
  __syncthreads();

  // exp + per-head sum
  float lsum[H_];
#pragma unroll
  for (int g = 0; g < H_; ++g) lsum[g] = 0.f;
  for (int j = 0; j < 4; ++j) {
    const int k = tid * 4 + j;
#pragma unroll
    for (int g = 0; g < H_; ++g) {
      float e = __expf(rowbuf[g * S_ + k] - rmax[g]);
      rowbuf[g * S_ + k] = e;
      lsum[g] += e;
    }
  }
#pragma unroll
  for (int g = 0; g < H_; ++g) red[g * 256 + tid] = lsum[g];
  for (int s2 = 128; s2 > 0; s2 >>= 1) {
    __syncthreads();
    if (tid < s2)
#pragma unroll
      for (int g = 0; g < H_; ++g)
        red[g * 256 + tid] += red[g * 256 + tid + s2];
  }
  __syncthreads();
  float rinv[H_];
#pragma unroll
  for (int g = 0; g < H_; ++g) rinv[g] = 1.0f / red[g * 256];

  // post-softmax head mix, bf16 output (4 contiguous k per thread -> b64 stores)
  const int k0 = tid * 4;
  v4bf ov[H_];
  for (int j = 0; j < 4; ++j) {
    float p[H_];
#pragma unroll
    for (int h = 0; h < H_; ++h) p[h] = rowbuf[h * S_ + k0 + j] * rinv[h];
#pragma unroll
    for (int g = 0; g < H_; ++g) {
      float m = bw_s[g];
#pragma unroll
      for (int h = 0; h < H_; ++h) m = fmaf(p[h], ww_s[g * 12 + h], m);
      ov[g][j] = f2bf(m);
    }
  }
#pragma unroll
  for (int g = 0; g < H_; ++g) {
    size_t o = ((size_t)(b * H_ + g) * S_ + qrow) * S_ + k0;
    *(v4bf*)(attn2 + o) = ov[g];
  }
}

// ---------- K4: v [bh][s][e] -> vT [bh][e][s] ----------

__global__ __launch_bounds__(256) void transpose_v_kernel(const bf16_t* __restrict__ v,
                                                          bf16_t* __restrict__ vT) {
  int idx = blockIdx.x * 256 + threadIdx.x;      // BH*E_*S_ total
  int bh = idx >> 16;
  int r  = idx & 65535;
  int e = r >> 10, s = r & 1023;
  vT[idx] = v[((size_t)bh << 16) + (s << 6) + e];
}

// ---------- K5: out = attn2 @ v per (b,h) ----------
// jobs = 48 bh x 64 q-tiles (full E per wave) = 3072.

__global__ __launch_bounds__(128) void av_gemm_kernel(
    const bf16_t* __restrict__ attn2, const bf16_t* __restrict__ vT,
    bf16_t* __restrict__ outp) {
  const int lane = threadIdx.x & 31;
  const int job  = blockIdx.x * 4 + (threadIdx.x >> 5);
  const int bh = job >> 6;
  const int tq = job & 63;
  const bf16_t* Abase = attn2 + ((size_t)bh * S_ + tq * 16) * S_;
  const bf16_t* Bb    = vT + (size_t)bh * E_ * S_;
  v8f acc[4] = {};
  for (int kk = 0; kk < S_; kk += 32) {
    v16bf a  = load_a_frag(Abase, S_, lane, kk);
    v16bf b0 = load_b_frag(Bb +  0 * S_, S_, lane, kk);
    v16bf b1 = load_b_frag(Bb + 16 * S_, S_, lane, kk);
    v16bf b2 = load_b_frag(Bb + 32 * S_, S_, lane, kk);
    v16bf b3 = load_b_frag(Bb + 48 * S_, S_, lane, kk);
    acc[0] = wmma_bf16(a, b0, acc[0]);
    acc[1] = wmma_bf16(a, b1, acc[1]);
    acc[2] = wmma_bf16(a, b2, acc[2]);
    acc[3] = wmma_bf16(a, b3, acc[3]);
  }
  const int n = lane & 15, mh = (lane >> 4) << 3;
  const int b = bh / 12, h = bh % 12;
#pragma unroll
  for (int i = 0; i < 4; ++i)
#pragma unroll
    for (int r = 0; r < 8; ++r) {
      int qq = tq * 16 + mh + r;
      outp[(((size_t)b * S_ + qq) * H_ + h) * E_ + i * 16 + n] = f2bf(acc[i][r]);
    }
}

// ---------- K6: y = out @ Wp^T + bp ----------
// jobs = 256 m-tiles x 12 n-groups = 3072.

__global__ __launch_bounds__(128) void proj_gemm_kernel(
    const bf16_t* __restrict__ outp, const bf16_t* __restrict__ wpb,
    const float* __restrict__ bp, float* __restrict__ y) {
  const int lane = threadIdx.x & 31;
  const int job  = blockIdx.x * 4 + (threadIdx.x >> 5);
  const int tm  = job & 255;
  const int tn4 = job >> 8;
  const bf16_t* Abase = outp + (size_t)(tm * 16) * D_;
  const bf16_t* Bb    = wpb + (size_t)(tn4 * 64) * D_;
  v8f acc[4] = {};
  for (int kk = 0; kk < D_; kk += 32) {
    v16bf a  = load_a_frag(Abase, D_, lane, kk);
    v16bf b0 = load_b_frag(Bb +  0 * D_, D_, lane, kk);
    v16bf b1 = load_b_frag(Bb + 16 * D_, D_, lane, kk);
    v16bf b2 = load_b_frag(Bb + 32 * D_, D_, lane, kk);
    v16bf b3 = load_b_frag(Bb + 48 * D_, D_, lane, kk);
    acc[0] = wmma_bf16(a, b0, acc[0]);
    acc[1] = wmma_bf16(a, b1, acc[1]);
    acc[2] = wmma_bf16(a, b2, acc[2]);
    acc[3] = wmma_bf16(a, b3, acc[3]);
  }
  const int n = lane & 15, mh = (lane >> 4) << 3;
#pragma unroll
  for (int i = 0; i < 4; ++i)
#pragma unroll
    for (int r = 0; r < 8; ++r) {
      int m = tm * 16 + mh + r;
      int f = tn4 * 64 + i * 16 + n;
      y[(size_t)m * D_ + f] = acc[i][r] + bp[f];
    }
}

// ---------- launch ----------

extern "C" void kernel_launch(void* const* d_in, const int* in_sizes, int n_in,
                              void* d_out, int out_size, void* d_ws, size_t ws_size,
                              hipStream_t stream) {
  (void)in_sizes; (void)n_in; (void)out_size; (void)ws_size;
  const float* x    = (const float*)d_in[0];
  const float* Wqkv = (const float*)d_in[1];
  const float* Wl   = (const float*)d_in[2];
  const float* bl   = (const float*)d_in[3];
  const float* Ww   = (const float*)d_in[4];
  const float* bw   = (const float*)d_in[5];
  const float* Wp   = (const float*)d_in[6];
  const float* bp   = (const float*)d_in[7];
  float* y = (float*)d_out;

  char* ws = (char*)d_ws;
  size_t off = 0;
  auto alloc = [&](size_t bytes) -> char* {
    char* p = ws + off;
    off += (bytes + 255) & ~(size_t)255;
    return p;
  };
  bf16_t* xb    = (bf16_t*)alloc((size_t)BS * D_ * 2);       // 6 MB
  bf16_t* wqkvb = (bf16_t*)alloc((size_t)D3 * D_ * 2);       // 3.5 MB
  bf16_t* wpb   = (bf16_t*)alloc((size_t)D_ * D_ * 2);       // 1.2 MB
  bf16_t* qb    = (bf16_t*)alloc((size_t)BH * S_ * E_ * 2);  // 6 MB
  bf16_t* kb    = (bf16_t*)alloc((size_t)BH * S_ * E_ * 2);  // 6 MB
  bf16_t* vb    = (bf16_t*)alloc((size_t)BH * S_ * E_ * 2);  // 6 MB
  bf16_t* vT    = (bf16_t*)alloc((size_t)BH * E_ * S_ * 2);  // 6 MB
  bf16_t* outp  = (bf16_t*)alloc((size_t)BS * D_ * 2);       // 6 MB
  bf16_t* attn2 = (bf16_t*)alloc((size_t)BH * S_ * S_ * 2);  // 96 MB
  float*  attn  = (float*)alloc((size_t)BH * S_ * S_ * 4);   // 192 MB

  convert_kernel<<<(BS * D_ + 255) / 256, 256, 0, stream>>>(x, xb, BS * D_);
  convert_kernel<<<(D3 * D_ + 255) / 256, 256, 0, stream>>>(Wqkv, wqkvb, D3 * D_);
  convert_kernel<<<(D_ * D_ + 255) / 256, 256, 0, stream>>>(Wp, wpb, D_ * D_);

  qkv_gemm_kernel<<<2304, 128, 0, stream>>>(xb, wqkvb, qb, kb, vb);
  attn_gemm_kernel<<<12288, 128, 0, stream>>>(qb, kb, attn);
  transpose_v_kernel<<<12288, 256, 0, stream>>>(vb, vT);
  th_softmax_kernel<<<BS, 256, 0, stream>>>(attn, Wl, bl, Ww, bw, attn2);
  av_gemm_kernel<<<768, 128, 0, stream>>>(attn2, vT, outp);
  proj_gemm_kernel<<<768, 128, 0, stream>>>(outp, wpb, bp, y);
}
